// BiAttentionClassifier_57775900065974
// MI455X (gfx1250) — matrix operand order repaired and checked by reading, not compiled, not run on hardware
//
#include <hip/hip_runtime.h>
#include <hip/hip_bf16.h>

// ---------- types ----------
typedef __attribute__((ext_vector_type(16))) __bf16 v16bf;
typedef __attribute__((ext_vector_type(8)))  float  v8f;
typedef unsigned short u16;

union V16 { v16bf bf; uint4 u[2]; };

#define S_LEN 2048
#define H_DIM 1024
#define D_IN  512
#define NCLS  16
#define NBATCH 8
#define CHUNK 128
#define NCHUNK (S_LEN / CHUNK)
#define QROWS 32              // query rows per block (2 row-tiles of 16)

// ---------- helpers ----------
__device__ __forceinline__ u16 f32_to_bf16(float f) {
  union { float f; unsigned u; } v; v.f = f;
  unsigned r = v.u + 0x7FFFu + ((v.u >> 16) & 1u);
  return (u16)(r >> 16);
}

// A-matrix 16x32 bf16 layout (wave32):
//  lanes 0-15 : M=lane,    elems 0-15 = K[0..15]
//  lanes 16-31: M=lane-16, elems 0-15 = K[16..31]
// p points at row base + k0 (k0 multiple of 32); kb = (lane>>4)*8
__device__ __forceinline__ v16bf load_a16(const u16* p, int kb) {
  V16 r;
  r.u[0] = *(const uint4*)(p + 2 * kb);
  r.u[1] = *(const uint4*)(p + 2 * kb + 8);
  return r.bf;
}

// B-matrix 32x16 bf16 layout: lane n holds column (n&15), K = (lane>>4)*16 + 0..15
// p points at &col_data[k0 + (lane>>4)*16], 16 contiguous bf16
__device__ __forceinline__ v16bf load_b16(const u16* p) {
  V16 r;
  r.u[0] = *(const uint4*)(p);
  r.u[1] = *(const uint4*)(p + 8);
  return r.bf;
}

__device__ __forceinline__ v8f wmma_bf16(v16bf a, v16bf b, v8f c) {
  return __builtin_amdgcn_wmma_f32_16x16x32_bf16(false, a, false, b, (short)0, c,
                                                 false, false);
}

// Direct global -> LDS async copy of one 16-byte chunk (no VGPR round trip).
// gfx1250: tracked with ASYNCcnt. lds_off = LDS byte address (low 32 bits of
// the flat shared address), gaddr = 64-bit global address.
__device__ __forceinline__ void async_g2lds_b128(unsigned lds_off,
                                                 unsigned long long gaddr) {
  asm volatile("global_load_async_to_lds_b128 %0, %1, off"
               :: "v"(lds_off), "v"(gaddr)
               : "memory");
}
__device__ __forceinline__ void wait_asynccnt0() {
  asm volatile("s_wait_asynccnt 0x0" ::: "memory");
}

// ---------- kernel 0: f32 -> bf16 convert ----------
__global__ void convert_f32_bf16(const float* __restrict__ src,
                                 u16* __restrict__ dst, int n) {
  int i = blockIdx.x * blockDim.x + threadIdx.x;
  if (i < n) dst[i] = f32_to_bf16(src[i]);
}

// ---------- kernel 1: r = x @ W1^T + b1 ----------
// grid (M/128, H/128), 256 threads. Wave w owns rows [m0+16w, +16) x 128 cols.
__global__ __launch_bounds__(256)
void gemm1_kernel(const u16* __restrict__ xb,   // [M, 512] bf16
                  const u16* __restrict__ w1b,  // [1024, 512] bf16
                  const float* __restrict__ b1, // [1024]
                  float* __restrict__ r_f32,    // [M, 1024]
                  u16* __restrict__ r_b16,      // [M, 1024]
                  u16* __restrict__ rT_b16) {   // [B, 1024, 2048]
  const int tid = threadIdx.x, wave = tid >> 5, lane = tid & 31;
  const int hf = lane >> 4, l15 = lane & 15;
  const int m0 = blockIdx.x * 128 + wave * 16;
  const int n0 = blockIdx.y * 128;
  const int kbA = hf * 8, kbB = hf * 16;

  v8f acc[8] = {};

  const u16* arow = xb + (long long)(m0 + l15) * D_IN;
  for (int k0 = 0; k0 < D_IN; k0 += 32) {
    v16bf a = load_a16(arow + k0, kbA);
#pragma unroll
    for (int t = 0; t < 8; ++t) {
      const u16* brow = w1b + (long long)(n0 + t * 16 + l15) * D_IN + k0 + kbB;
      acc[t] = wmma_bf16(a, load_b16(brow), acc[t]);
    }
  }

  const int bidx = m0 / S_LEN;   // M-tiles of 128 never straddle a batch
  const int s0 = m0 % S_LEN;
#pragma unroll
  for (int t = 0; t < 8; ++t) {
    int col = n0 + t * 16 + l15;
    float bias = b1[col];
    union { u16 s[8]; uint4 u; } pk;
#pragma unroll
    for (int v = 0; v < 8; ++v) {
      float val = acc[t][v] + bias;
      long long idx = (long long)(m0 + v + 8 * hf) * H_DIM + col;
      r_f32[idx] = val;
      u16 bv = f32_to_bf16(val);
      r_b16[idx] = bv;
      pk.s[v] = bv;   // rows m = v + 8*hf are contiguous in s
    }
    *(uint4*)(rT_b16 + ((long long)bidx * H_DIM + col) * S_LEN + s0 + 8 * hf) = pk.u;
  }
}

// ---------- kernel 2: flash attention + residual + LayerNorm ----------
// grid (S/QROWS, B), 256 threads (8 waves). Wave w owns features [128w, +128).
// M=32 query rows per block: each B operand (K rows / V cols) feeds 2 WMMAs,
// halving L2 operand traffic vs M=16.
struct AttnSmem {
  u16   Qs[QROWS * H_DIM];     // 64 KB
  float Ss[QROWS * CHUNK];     // 16 KB
  u16   Ps[QROWS * CHUNK];     // 8 KB
  float pmax[QROWS][8];
  float psum[QROWS][8];
  float m[QROWS], l[QROWS], factor[QROWS], linv[QROWS];
  float sum[QROWS], sumsq[QROWS], mu[QROWS], rstd[QROWS];
};

__global__ __launch_bounds__(256)
void attn_kernel(const u16* __restrict__ r_b16,   // [B*S, H]
                 const u16* __restrict__ rT_b16,  // [B, H, S]
                 const float* __restrict__ r_f32, // [B*S, H]
                 const float* __restrict__ gamma,
                 const float* __restrict__ beta,
                 u16* __restrict__ y_b16) {       // [B*S, H]
  extern __shared__ char smem_raw[];
  AttnSmem& sm = *(AttnSmem*)smem_raw;

  const int tid = threadIdx.x, wave = tid >> 5, lane = tid & 31;
  const int hf = lane >> 4, l15 = lane & 15;
  const int b = blockIdx.y;
  const long long qrow0 = (long long)b * S_LEN + blockIdx.x * QROWS;
  const int kbA = hf * 8, kbB = hf * 16;

  { // Q tile (32 x 1024 bf16 = 64 KB) via direct global->LDS async DMA
    const char* src = (const char*)(r_b16 + qrow0 * H_DIM);
    for (int i = tid; i < (QROWS * H_DIM * 2) / 16; i += 256) {
      unsigned lds_off =
          (unsigned)(unsigned long long)(uintptr_t)((char*)sm.Qs + i * 16);
      unsigned long long gaddr =
          (unsigned long long)(uintptr_t)(src + i * 16);
      async_g2lds_b128(lds_off, gaddr);
    }
    wait_asynccnt0();
  }
  if (tid < QROWS) { sm.m[tid] = -1e30f; sm.l[tid] = 0.f; }
  __syncthreads();

  v8f acc[2][8] = {};   // 32 rows x 128 feats fp32 (2 row-tiles x 8 col-tiles)

  for (int c = 0; c < NCHUNK; ++c) {
    const int key0 = c * CHUNK;

    // prefetch next chunk's K rows / V columns into cache while this chunk's
    // softmax phase runs (L2-resident working set; hides latency)
    if (c + 1 < NCHUNK) {
      const u16* nk =
          r_b16 + ((long long)b * S_LEN + key0 + CHUNK + wave * 16 + l15) * H_DIM;
      __builtin_prefetch(nk, 0, 1);
      const u16* nv =
          rT_b16 + ((long long)b * H_DIM + wave * 128 + l15) * S_LEN + key0 + CHUNK;
      __builtin_prefetch(nv, 0, 1);
    }

    // scores: wave w computes keys [key0+16w, +16) vs 32 query rows;
    // one B load feeds both query row-tiles.
    v8f sacc[2] = {};
    const u16* krow =
        r_b16 + ((long long)b * S_LEN + key0 + wave * 16 + l15) * H_DIM;
    for (int k0 = 0; k0 < H_DIM; k0 += 32) {
      v16bf bb = load_b16(krow + k0 + kbB);
#pragma unroll
      for (int q = 0; q < 2; ++q) {
        v16bf a = load_a16(sm.Qs + (16 * q + l15) * H_DIM + k0, kbA);
        sacc[q] = wmma_bf16(a, bb, sacc[q]);
      }
    }
#pragma unroll
    for (int q = 0; q < 2; ++q)
#pragma unroll
      for (int v = 0; v < 8; ++v)
        sm.Ss[(16 * q + v + 8 * hf) * CHUNK + wave * 16 + l15] = sacc[q][v];
    __syncthreads();

    // online softmax over this 32x128 chunk (8 groups of 16 cols per row)
    {
      int row = tid >> 3, grp = tid & 7;
      const float* srow = sm.Ss + row * CHUNK + grp * 16;
      float mx = srow[0];
#pragma unroll
      for (int i = 1; i < 16; ++i) mx = fmaxf(mx, srow[i]);
      sm.pmax[row][grp] = mx;
    }
    __syncthreads();
    if (tid < QROWS) {
      float mx = sm.pmax[tid][0];
      for (int i = 1; i < 8; ++i) mx = fmaxf(mx, sm.pmax[tid][i]);
      float m_old = sm.m[tid], m_new = fmaxf(m_old, mx);
      float f = __expf(m_old - m_new);
      sm.factor[tid] = f;
      sm.m[tid] = m_new;
      sm.l[tid] *= f;
    }
    __syncthreads();
    {
      int row = tid >> 3, grp = tid & 7;
      float m_new = sm.m[row];
      const float* srow = sm.Ss + row * CHUNK + grp * 16;
      u16* prow = sm.Ps + row * CHUNK + grp * 16;
      float ssum = 0.f;
#pragma unroll
      for (int i = 0; i < 16; ++i) {
        float e = __expf(srow[i] - m_new);
        ssum += e;
        prow[i] = f32_to_bf16(e);
      }
      sm.psum[row][grp] = ssum;
    }
    __syncthreads();
    if (tid < QROWS) {
      float s = 0.f;
      for (int i = 0; i < 8; ++i) s += sm.psum[tid][i];
      sm.l[tid] += s;
    }
    __syncthreads();

    { // rescale running accumulators
#pragma unroll
      for (int q = 0; q < 2; ++q) {
        float f0[8];
#pragma unroll
        for (int v = 0; v < 8; ++v) f0[v] = sm.factor[16 * q + v + 8 * hf];
#pragma unroll
        for (int t = 0; t < 8; ++t)
#pragma unroll
          for (int v = 0; v < 8; ++v) acc[q][t][v] *= f0[v];
      }
    }

    // P·V: K = 128 chunk keys; each V-column B load feeds 2 WMMAs.
    for (int k0 = 0; k0 < CHUNK; k0 += 32) {
      v16bf a0 = load_a16(sm.Ps + l15 * CHUNK + k0, kbA);
      v16bf a1 = load_a16(sm.Ps + (16 + l15) * CHUNK + k0, kbA);
#pragma unroll
      for (int t = 0; t < 8; ++t) {
        int feat = wave * 128 + t * 16 + l15;
        const u16* vp =
            rT_b16 + ((long long)b * H_DIM + feat) * S_LEN + key0 + k0 + kbB;
        v16bf bb = load_b16(vp);
        acc[0][t] = wmma_bf16(a0, bb, acc[0][t]);
        acc[1][t] = wmma_bf16(a1, bb, acc[1][t]);
      }
    }
    __syncthreads();   // protect Ss/Ps for next chunk
  }

  // epilogue: 1/l scale + residual (in place), LayerNorm, emit bf16 y
  if (tid < QROWS) {
    sm.linv[tid] = 1.0f / sm.l[tid];
    sm.sum[tid] = 0.f;
    sm.sumsq[tid] = 0.f;
  }
  __syncthreads();

#pragma unroll
  for (int q = 0; q < 2; ++q) {
    float li[8];
#pragma unroll
    for (int v = 0; v < 8; ++v) li[v] = sm.linv[16 * q + v + 8 * hf];
#pragma unroll
    for (int t = 0; t < 8; ++t) {
      int feat = wave * 128 + t * 16 + l15;
#pragma unroll
      for (int v = 0; v < 8; ++v) {
        long long m = qrow0 + 16 * q + v + 8 * hf;
        acc[q][t][v] = acc[q][t][v] * li[v] + r_f32[m * H_DIM + feat];
      }
    }
  }
#pragma unroll
  for (int q = 0; q < 2; ++q)
#pragma unroll
    for (int v = 0; v < 8; ++v) {
      float s = 0.f, s2 = 0.f;
#pragma unroll
      for (int t = 0; t < 8; ++t) { float x = acc[q][t][v]; s += x; s2 += x * x; }
      atomicAdd(&sm.sum[16 * q + v + 8 * hf], s);
      atomicAdd(&sm.sumsq[16 * q + v + 8 * hf], s2);
    }
  __syncthreads();
  if (tid < QROWS) {
    float mu = sm.sum[tid] * (1.0f / H_DIM);
    float var = sm.sumsq[tid] * (1.0f / H_DIM) - mu * mu;
    sm.mu[tid] = mu;
    sm.rstd[tid] = rsqrtf(var + 1e-5f);
  }
  __syncthreads();
#pragma unroll
  for (int q = 0; q < 2; ++q) {
    float mu[8], rs[8];
#pragma unroll
    for (int v = 0; v < 8; ++v) {
      mu[v] = sm.mu[16 * q + v + 8 * hf];
      rs[v] = sm.rstd[16 * q + v + 8 * hf];
    }
#pragma unroll
    for (int t = 0; t < 8; ++t) {
      int feat = wave * 128 + t * 16 + l15;
      float g = gamma[feat], be = beta[feat];
#pragma unroll
      for (int v = 0; v < 8; ++v) {
        float y = (acc[q][t][v] - mu[v]) * rs[v] * g + be;
        y_b16[(qrow0 + 16 * q + v + 8 * hf) * H_DIM + feat] = f32_to_bf16(y);
      }
    }
  }
}

// ---------- kernel 3: out = y @ W2^T + b2 ----------
// grid (M/(16*8)), 256 threads; wave w handles one 16-row tile, N = 16.
__global__ __launch_bounds__(256)
void gemm2_kernel(const u16* __restrict__ yb,   // [M, 1024]
                  const u16* __restrict__ w2b,  // [16, 1024]
                  const float* __restrict__ b2, // [16]
                  float* __restrict__ out) {    // [M, 16]
  const int tid = threadIdx.x, wave = tid >> 5, lane = tid & 31;
  const int hf = lane >> 4, l15 = lane & 15;
  const int m0 = (blockIdx.x * 8 + wave) * 16;
  const int kbA = hf * 8, kbB = hf * 16;

  v8f acc = {};
  const u16* arow = yb + (long long)(m0 + l15) * H_DIM;
  const u16* brow = w2b + (long long)l15 * H_DIM;
  for (int k0 = 0; k0 < H_DIM; k0 += 32)
    acc = wmma_bf16(load_a16(arow + k0, kbA), load_b16(brow + k0 + kbB), acc);

  float bias = b2[l15];
#pragma unroll
  for (int v = 0; v < 8; ++v)
    out[(long long)(m0 + v + 8 * hf) * NCLS + l15] = acc[v] + bias;
}

// ---------- launcher ----------
extern "C" void kernel_launch(void* const* d_in, const int* in_sizes, int n_in,
                              void* d_out, int out_size, void* d_ws, size_t ws_size,
                              hipStream_t stream) {
  const float* x     = (const float*)d_in[0];  // [8,2048,512]
  const float* W1    = (const float*)d_in[1];  // [1024,512]
  const float* b1    = (const float*)d_in[2];  // [1024]
  const float* gamma = (const float*)d_in[3];  // [1024]
  const float* beta  = (const float*)d_in[4];  // [1024]
  const float* W2    = (const float*)d_in[5];  // [16,1024]
  const float* b2    = (const float*)d_in[6];  // [16]
  float* out = (float*)d_out;                  // [8,2048,16]

  const long long M = (long long)NBATCH * S_LEN;          // 16384
  const long long nx = M * D_IN;                          // 8,388,608
  const long long nw1 = (long long)H_DIM * D_IN;          // 524,288
  const long long nw2 = (long long)NCLS * H_DIM;          // 16,384

  char* ws = (char*)d_ws;
  size_t off = 0;
  auto carve = [&](size_t bytes) {
    size_t o = off;
    off = (off + bytes + 255) & ~(size_t)255;
    return o;
  };
  u16* xb      = (u16*)(ws + carve(nx * 2));
  u16* w1b     = (u16*)(ws + carve(nw1 * 2));
  u16* w2b     = (u16*)(ws + carve(nw2 * 2));
  float* r_f32 = (float*)(ws + carve(M * H_DIM * 4));
  u16* r_b16   = (u16*)(ws + carve(M * H_DIM * 2));
  u16* rT_b16  = (u16*)(ws + carve(M * H_DIM * 2));
  u16* y_b16   = (u16*)(ws + carve(M * H_DIM * 2));
  (void)ws_size; (void)n_in; (void)in_sizes; (void)out_size;

  // 0) converts
  convert_f32_bf16<<<(unsigned)((nx + 255) / 256), 256, 0, stream>>>(x, xb, (int)nx);
  convert_f32_bf16<<<(unsigned)((nw1 + 255) / 256), 256, 0, stream>>>(W1, w1b, (int)nw1);
  convert_f32_bf16<<<(unsigned)((nw2 + 255) / 256), 256, 0, stream>>>(W2, w2b, (int)nw2);

  // 1) r = x W1^T + b1  (also emits bf16 and transposed-bf16 copies)
  gemm1_kernel<<<dim3((unsigned)(M / 128), H_DIM / 128), 256, 0, stream>>>(
      xb, w1b, b1, r_f32, r_b16, rT_b16);

  // 2) flash attention + residual + LayerNorm -> y (bf16); ~93 KB dynamic LDS
  attn_kernel<<<dim3(S_LEN / QROWS, NBATCH), 256, sizeof(AttnSmem), stream>>>(
      r_b16, rT_b16, r_f32, gamma, beta, y_b16);

  // 3) out = y W2^T + b2
  gemm2_kernel<<<(unsigned)(M / (16 * 8)), 256, 0, stream>>>(y_b16, w2b, b2, out);
}
